// SpatioTemporalAttentionGNN_10058813407712
// MI455X (gfx1250) — matrix operand order repaired and checked by reading, not compiled, not run on hardware
//
#include <hip/hip_runtime.h>
#include <hip/hip_bf16.h>
#include <math.h>

// ---------------- problem constants (from reference) ----------------
#define NND   2000          // nodes
#define NFT   16            // input features
#define HD    64            // hidden
#define BBT   4             // batch
#define SST   96            // seq len
#define EET   16000         // edges
#define SNK   (SST*NND)     // 192000 keys per batch
#define BSNK  (BBT*SNK)     // 768000 total (b,s,n) rows
#define OWN   24            // output window
#define P2C   (NND*HD)      // 128000 pre2 output cols
#define CTXBLK 240
#define CHUNK  (SNK/CTXBLK) // 800

typedef __attribute__((ext_vector_type(16))) _Float16 v16h;
typedef __attribute__((ext_vector_type(8)))  _Float16 v8h;
typedef __attribute__((ext_vector_type(2)))  _Float16 v2h;
typedef __attribute__((ext_vector_type(8)))  float    v8f;

__device__ __forceinline__ float eluf(float x){ return x > 0.f ? x : __expf(x) - 1.f; }
__device__ __forceinline__ float softplusf(float x){ return x > 20.f ? x : log1pf(__expf(x)); }
__device__ __forceinline__ float sigmoidf(float x){ return 1.f/(1.f+__expf(-x)); }

// ---------------- graph preprocessing (deterministic, no float atomics) ----
__global__ void k_graph_deg(const int* __restrict__ ei, const float* __restrict__ ew,
                            float* __restrict__ dinv, int* __restrict__ counts){
  int n = blockIdx.x*blockDim.x + threadIdx.x;
  if (n >= NND) return;
  const int* dst = ei + EET;
  float deg = 1.0f; int cnt = 1;                 // self loop contributes 1.0
  for (int e = 0; e < EET; ++e)
    if (dst[e] == n){ deg += ew[e]; ++cnt; }
  dinv[n]   = deg > 0.f ? rsqrtf(deg) : 0.f;
  counts[n] = cnt;
}

__global__ void k_scan(const int* __restrict__ counts, int* __restrict__ row_ptr){
  if (blockIdx.x==0 && threadIdx.x==0){
    int acc = 0;
    for (int n=0;n<NND;++n){ row_ptr[n]=acc; acc+=counts[n]; }
    row_ptr[NND]=acc;
  }
}

__global__ void k_csr_fill(const int* __restrict__ ei, const float* __restrict__ ew,
                           const float* __restrict__ dinv, const int* __restrict__ row_ptr,
                           int* __restrict__ csr_src, float* __restrict__ csr_coef){
  int n = blockIdx.x*blockDim.x + threadIdx.x;
  if (n >= NND) return;
  const int* src = ei; const int* dst = ei + EET;
  int   p  = row_ptr[n];
  float dn = dinv[n];
  for (int e = 0; e < EET; ++e){
    if (dst[e] == n){
      int s = src[e];
      csr_src[p]  = s;
      csr_coef[p] = dinv[s]*ew[e]*dn;
      ++p;
    }
  }
  csr_src[p]  = n;          // self loop last (matches jnp.concatenate order)
  csr_coef[p] = dn*dn;
}

// ---------------- fused GCN:  h = elu(W * (sum coef*x_src) + b) -> f16 -----
__global__ void __launch_bounds__(256) k_gcn(const float* __restrict__ x,
    const float* __restrict__ Wg, const float* __restrict__ bg,
    const int* __restrict__ row_ptr, const int* __restrict__ csr_src,
    const float* __restrict__ csr_coef, _Float16* __restrict__ h){
  int lane = threadIdx.x & 31;
  int wave = (blockIdx.x*blockDim.x + threadIdx.x) >> 5;   // 0..95999
  int c0 = lane*2;
  float w0[NFT], w1[NFT];
#pragma unroll
  for (int f=0; f<NFT; ++f){ w0[f]=Wg[c0*NFT+f]; w1[f]=Wg[(c0+1)*NFT+f]; }
  float bg0 = bg[c0], bg1 = bg[c0+1];
  for (int i=0;i<8;++i){
    int item = wave + i*96000;                 // 0..767999
    int n  = item % NND;
    int bs = item / NND;
    int p0 = row_ptr[n], p1 = row_ptr[n+1];
    float xa[NFT];
#pragma unroll
    for (int f=0;f<NFT;++f) xa[f]=0.f;
    const float* xbase = x + (size_t)bs*NND*NFT;
    for (int p=p0; p<p1; ++p){
      int   s  = csr_src[p];
      float cf = csr_coef[p];
      const float4* xp = (const float4*)(xbase + (size_t)s*NFT);
#pragma unroll
      for (int q=0;q<4;++q){
        float4 v = xp[q];
        xa[4*q+0]+=cf*v.x; xa[4*q+1]+=cf*v.y; xa[4*q+2]+=cf*v.z; xa[4*q+3]+=cf*v.w;
      }
    }
    float m0=0.f, m1=0.f;
#pragma unroll
    for (int f=0;f<NFT;++f){ m0 += w0[f]*xa[f]; m1 += w1[f]*xa[f]; }
    v2h out; out.x = (_Float16)eluf(m0+bg0); out.y = (_Float16)eluf(m1+bg1);
    *(v2h*)(h + (size_t)item*HD + c0) = out;
  }
}

// ---------------- query = mean_n h[b,S-1,n,:]; qp = Wq * query -------------
__global__ void k_query(const _Float16* __restrict__ h, const float* __restrict__ Wq,
                        float* __restrict__ qp){
  int b = blockIdx.x, t = threadIdx.x;
  int c = t & 63, part = t >> 6;                 // 512 threads: 8 parts x 64 cols
  __shared__ float red[8][64];
  __shared__ float q[64];
  const _Float16* hb = h + ((size_t)b*SNK + (size_t)(SST-1)*NND)*HD;
  float s=0.f;
  for (int n=part; n<NND; n+=8) s += (float)hb[(size_t)n*HD + c];
  red[part][c]=s; __syncthreads();
  if (t<64){ float a=0; for (int p=0;p<8;++p) a+=red[p][t]; q[t]=a*(1.0f/NND); }
  __syncthreads();
  if (t<64){ float a=0; for (int j=0;j<HD;++j) a+=Wq[t*HD+j]*q[j]; qp[b*HD+t]=a; }
}

// ---------------- WMMA attention scores: v_w . tanh(qp + h*Wk^T) + v_b -----
__global__ void __launch_bounds__(256) k_scores(const _Float16* __restrict__ h,
    const float* __restrict__ Wk, const float* __restrict__ vw, const float* __restrict__ vb,
    const float* __restrict__ qp, float* __restrict__ scores){
  int lane = threadIdx.x & 31;
  int grp  = (blockIdx.x*blockDim.x + threadIdx.x) >> 5;   // 0..47999
  int b = grp / (SNK/16);
  int g = grp % (SNK/16);
  int n16 = lane & 15;
  int hiK = (lane & 16) ? 1 : 0;

  // B fragments = Wk^T tiles (B[k][n] = Wk[n][k]), 4 N-tiles x 2 K-chunks
  v16h bf[8];
#pragma unroll
  for (int t=0;t<4;++t){
#pragma unroll
    for (int c=0;c<2;++c){
      const float4* wp = (const float4*)(Wk + (t*16+n16)*HD + c*32 + hiK*16);
      v16h r;
#pragma unroll
      for (int q=0;q<4;++q){
        float4 v = wp[q];
        r[4*q+0]=(_Float16)v.x; r[4*q+1]=(_Float16)v.y;
        r[4*q+2]=(_Float16)v.z; r[4*q+3]=(_Float16)v.w;
      }
      bf[t*2+c]=r;
    }
  }
  float qpv[4], vwv[4];
#pragma unroll
  for (int t=0;t<4;++t){ qpv[t]=qp[b*HD+t*16+n16]; vwv[t]=vw[t*16+n16]; }

  v8f acc[4];
#pragma unroll
  for (int t=0;t<4;++t)
#pragma unroll
    for (int q=0;q<8;++q) acc[t][q]=0.f;

  size_t rowbase = ((size_t)b*SNK + (size_t)g*16 + n16)*HD;
#pragma unroll
  for (int c=0;c<2;++c){
    const v8h* hp = (const v8h*)(h + rowbase + c*32 + hiK*16);
    v8h lo = hp[0];
    v8h hi = hp[1];                 // +8 halves
    v16h a;
#pragma unroll
    for (int q=0;q<8;++q){ a[q]=lo[q]; a[8+q]=hi[q]; }
#pragma unroll
    for (int t=0;t<4;++t)
      acc[t] = __builtin_amdgcn_wmma_f32_16x16x32_f16(
                 false, a, false, bf[t*2+c], (short)0, acc[t], false, false);
  }

  float vb0 = vb[0];
#pragma unroll
  for (int r=0;r<8;++r){
    float p = 0.f;
#pragma unroll
    for (int t=0;t<4;++t) p += vwv[t]*tanhf(qpv[t]+acc[t][r]);
    p += __shfl_xor(p,1,32); p += __shfl_xor(p,2,32);
    p += __shfl_xor(p,4,32); p += __shfl_xor(p,8,32);
    if (lane==0)  scores[(size_t)b*SNK + g*16 + r]     = p + vb0;
    if (lane==16) scores[(size_t)b*SNK + g*16 + 8 + r] = p + vb0;
  }
}

// ---------------- softmax over SN per batch (deterministic tree) -----------
__global__ void k_softmax(const float* __restrict__ scores, float* __restrict__ attn){
  int b = blockIdx.x, t = threadIdx.x;
  __shared__ float red[1024];
  const float* sc = scores + (size_t)b*SNK;
  float mx = -3.0e38f;
  for (int k=t; k<SNK; k+=1024) mx = fmaxf(mx, sc[k]);
  red[t]=mx; __syncthreads();
  for (int o=512;o>0;o>>=1){ if (t<o) red[t]=fmaxf(red[t],red[t+o]); __syncthreads(); }
  mx = red[0]; __syncthreads();
  float s=0.f;
  for (int k=t; k<SNK; k+=1024) s += __expf(sc[k]-mx);
  red[t]=s; __syncthreads();
  for (int o=512;o>0;o>>=1){ if (t<o) red[t]+=red[t+o]; __syncthreads(); }
  float inv = 1.0f/red[0];
  float* ab = attn + (size_t)b*SNK;
  for (int k=t; k<SNK; k+=1024) ab[k] = __expf(sc[k]-mx)*inv;
}

// ---------------- ctxh = sum_k attn_k * h_k  (block partials) --------------
__global__ void __launch_bounds__(256) k_ctxh(const _Float16* __restrict__ h,
    const float* __restrict__ attn, float* __restrict__ partials){
  int b = blockIdx.y, blk = blockIdx.x, t = threadIdx.x;
  int c = t & 63, r = t >> 6;
  __shared__ float red[4][64];
  size_t base = (size_t)b*SNK;
  float acc = 0.f;
  for (int i=0;i<CHUNK/4;++i){
    int k = blk*CHUNK + i*4 + r;
    acc += attn[base+k] * (float)h[(base+k)*HD + c];
  }
  red[r][c]=acc; __syncthreads();
  if (t<64)
    partials[((size_t)b*CTXBLK + blk)*HD + t] = red[0][t]+red[1][t]+red[2][t]+red[3][t];
}

// ---------------- tiny head: Wv matvec, LN1, GRU x2, LN2, pre1+elu ---------
__global__ void k_head(const float* __restrict__ partials, const float* __restrict__ Wv,
    const float* __restrict__ ln1g, const float* __restrict__ ln1b,
    const float* __restrict__ gWih, const float* __restrict__ gbih, const float* __restrict__ gbhh,
    const float* __restrict__ ln2g, const float* __restrict__ ln2b,
    const float* __restrict__ p1W, const float* __restrict__ p1b,
    _Float16* __restrict__ fbuf){
  int t = threadIdx.x;
  __shared__ float sA[64], sB[64], gi[192], stats[2];
  for (int b=0;b<BBT;++b){
    if (t<64){ float s=0; for (int p=0;p<CTXBLK;++p) s+=partials[((size_t)b*CTXBLK+p)*HD+t]; sA[t]=s; }
    __syncthreads();
    if (t<64){ float s=0; for (int j=0;j<HD;++j) s+=Wv[t*HD+j]*sA[j]; sB[t]=s; }   // context
    __syncthreads();
    if (t==0){
      float mu=0; for (int j=0;j<HD;++j) mu+=sB[j]; mu*=(1.f/HD);
      float var=0; for (int j=0;j<HD;++j){ float d=sB[j]-mu; var+=d*d; } var*=(1.f/HD);
      stats[0]=mu; stats[1]=rsqrtf(var+1e-5f);
    }
    __syncthreads();
    if (t<64) sA[t] = (sB[t]-stats[0])*stats[1]*ln1g[t] + ln1b[t];
    __syncthreads();
    for (int l=0;l<2;++l){
      if (t<192){ float s=gbih[l*192+t]; for (int j=0;j<HD;++j) s+=gWih[((size_t)l*192+t)*HD+j]*sA[j]; gi[t]=s; }
      __syncthreads();
      if (t<64){
        float r_ = sigmoidf(gi[t]       + gbhh[l*192+t]);
        float z_ = sigmoidf(gi[64+t]    + gbhh[l*192+64+t]);
        float n_ = tanhf  (gi[128+t] + r_*gbhh[l*192+128+t]);
        sB[t] = (1.f - z_)*n_;
      }
      __syncthreads();
      if (t<64) sA[t]=sB[t];
      __syncthreads();
    }
    if (t==0){
      float mu=0; for (int j=0;j<HD;++j) mu+=sA[j]; mu*=(1.f/HD);
      float var=0; for (int j=0;j<HD;++j){ float d=sA[j]-mu; var+=d*d; } var*=(1.f/HD);
      stats[0]=mu; stats[1]=rsqrtf(var+1e-5f);
    }
    __syncthreads();
    if (t<64) sB[t] = (sA[t]-stats[0])*stats[1]*ln2g[t] + ln2b[t];                 // final
    __syncthreads();
    if (t<128){
      float s = p1b[t];
      for (int j=0;j<HD;++j) s += p1W[t*HD+j]*sB[j];
      fbuf[b*128+t] = (_Float16)eluf(s);
    }
    __syncthreads();
  }
}

// ---------------- WMMA pre2: nf(4x128000) = f(4x128) @ pre2_W^T ------------
__global__ void __launch_bounds__(256) k_pre2(const _Float16* __restrict__ fbuf,
    const float* __restrict__ W2, const float* __restrict__ b2, float* __restrict__ nf){
  int lane = threadIdx.x & 31;
  int tile = (blockIdx.x*blockDim.x + threadIdx.x) >> 5;   // 0..7999
  int n16 = lane & 15;
  int hiK = (lane & 16) ? 1 : 0;

  v16h af[4];                                  // A (f rows, M padded 4->16)
#pragma unroll
  for (int c=0;c<4;++c){
    v16h a;
    if (n16 < BBT){
      const v8h* fp = (const v8h*)(fbuf + n16*128 + c*32 + hiK*16);
      v8h lo = fp[0], hi = fp[1];
#pragma unroll
      for (int q=0;q<8;++q){ a[q]=lo[q]; a[8+q]=hi[q]; }
    } else {
#pragma unroll
      for (int q=0;q<16;++q) a[q]=(_Float16)0.f;
    }
    af[c]=a;
  }

  int n0 = tile*16;
  v8f acc;
#pragma unroll
  for (int q=0;q<8;++q) acc[q]=0.f;
#pragma unroll
  for (int c=0;c<4;++c){
    const float4* wp = (const float4*)(W2 + (size_t)(n0+n16)*128 + c*32 + hiK*16);
    v16h bfv;
#pragma unroll
    for (int q=0;q<4;++q){
      float4 v = wp[q];
      bfv[4*q+0]=(_Float16)v.x; bfv[4*q+1]=(_Float16)v.y;
      bfv[4*q+2]=(_Float16)v.z; bfv[4*q+3]=(_Float16)v.w;
    }
    acc = __builtin_amdgcn_wmma_f32_16x16x32_f16(
            false, af[c], false, bfv, (short)0, acc, false, false);
  }
  if (tile + 1 < P2C/16)
    __builtin_prefetch(W2 + (size_t)(n0+16+n16)*128, 0, 1);   // global_prefetch_b8

  if (lane < 16){
    float bias = b2[n0 + lane];
#pragma unroll
    for (int r=0;r<BBT;++r)
      nf[(size_t)r*P2C + n0 + lane] = acc[r] + bias;
  }
}

// ---------------- monotone quantile head + transpose to (B,OW,N,1,3) -------
__global__ void __launch_bounds__(128) k_quant(const float* __restrict__ nf,
    const float* __restrict__ lW, const float* __restrict__ lb,
    const float* __restrict__ iW, const float* __restrict__ ib,
    float* __restrict__ out){
  int bn = blockIdx.x;
  int b = bn / NND, n = bn % NND;
  int t = threadIdx.x;
  __shared__ float row[64], ql[24], qi[48];
  if (t<64) row[t] = nf[(size_t)b*P2C + (size_t)n*HD + t];
  __syncthreads();
  if (t < 24){
    float s = lb[t];
    for (int j=0;j<HD;++j) s += lW[t*HD+j]*row[j];
    ql[t]=s;
  } else if (t < 72){
    int j2 = t-24;
    float s = ib[j2];
    for (int j=0;j<HD;++j) s += iW[j2*HD+j]*row[j];
    qi[j2]=s;
  }
  __syncthreads();
  if (t < 24){
    float q0 = ql[t];
    float q1 = q0 + softplusf(qi[2*t]);
    float q2 = q1 + softplusf(qi[2*t+1]);
    size_t base = (((size_t)b*OWN + t)*NND + n)*3;
    out[base]=q0; out[base+1]=q1; out[base+2]=q2;
  }
}

// ---------------- host side ------------------------------------------------
extern "C" void kernel_launch(void* const* d_in, const int* in_sizes, int n_in,
                              void* d_out, int out_size, void* d_ws, size_t ws_size,
                              hipStream_t stream){
  (void)in_sizes; (void)n_in; (void)out_size; (void)ws_size;
  const float* x   =(const float*)d_in[0];
  const int*   ei  =(const int*)  d_in[1];
  const float* ew  =(const float*)d_in[2];
  const float* Wg  =(const float*)d_in[3];
  const float* bg  =(const float*)d_in[4];
  const float* Wq  =(const float*)d_in[5];
  const float* Wk  =(const float*)d_in[6];
  const float* Wv  =(const float*)d_in[7];
  const float* vw  =(const float*)d_in[8];
  const float* vb  =(const float*)d_in[9];
  const float* ln1g=(const float*)d_in[10];
  const float* ln1b=(const float*)d_in[11];
  const float* gWih=(const float*)d_in[12];
  /* d_in[13] = gru_Whh unused: h_prev == 0 */
  const float* gbih=(const float*)d_in[14];
  const float* gbhh=(const float*)d_in[15];
  const float* ln2g=(const float*)d_in[16];
  const float* ln2b=(const float*)d_in[17];
  const float* p1W =(const float*)d_in[18];
  const float* p1b =(const float*)d_in[19];
  const float* p2W =(const float*)d_in[20];
  const float* p2b =(const float*)d_in[21];
  const float* lW  =(const float*)d_in[22];
  const float* lb  =(const float*)d_in[23];
  const float* iW  =(const float*)d_in[24];
  const float* ib  =(const float*)d_in[25];

  // workspace carve (~104 MB total)
  char* ws = (char*)d_ws;
  size_t off = 0;
  auto carve = [&](size_t bytes)->char*{
    char* p = ws + off;
    off = (off + bytes + 255) & ~(size_t)255;
    return p;
  };
  _Float16* hbuf    = (_Float16*)carve((size_t)BSNK*HD*sizeof(_Float16)); // 98.3 MB
  float*    scores  = (float*)   carve((size_t)BSNK*sizeof(float));       // 3.07 MB
  float*    dinv    = (float*)   carve(NND*sizeof(float));
  int*      counts  = (int*)     carve(NND*sizeof(int));
  int*      row_ptr = (int*)     carve((NND+1)*sizeof(int));
  int*      csr_src = (int*)     carve((EET+NND)*sizeof(int));
  float*    csr_cf  = (float*)   carve((EET+NND)*sizeof(float));
  float*    qp      = (float*)   carve(BBT*HD*sizeof(float));
  float*    partials= (float*)   carve((size_t)BBT*CTXBLK*HD*sizeof(float));
  _Float16* fbuf    = (_Float16*)carve(BBT*128*sizeof(_Float16));
  float*    nf      = (float*)   carve((size_t)BBT*P2C*sizeof(float));    // 2.05 MB

  float* pred = (float*)d_out;                      // (B,OW,N,1,3) = 576000
  float* attn = pred + (size_t)BBT*OWN*NND*3;       // (B,S,N)      = 768000

  k_graph_deg<<<8, 256, 0, stream>>>(ei, ew, dinv, counts);
  k_scan    <<<1, 1,   0, stream>>>(counts, row_ptr);
  k_csr_fill<<<8, 256, 0, stream>>>(ei, ew, dinv, row_ptr, csr_src, csr_cf);
  k_gcn     <<<12000, 256, 0, stream>>>(x, Wg, bg, row_ptr, csr_src, csr_cf, hbuf);
  k_query   <<<BBT, 512, 0, stream>>>(hbuf, Wq, qp);
  k_scores  <<<6000, 256, 0, stream>>>(hbuf, Wk, vw, vb, qp, scores);
  k_softmax <<<BBT, 1024, 0, stream>>>(scores, attn);
  k_ctxh    <<<dim3(CTXBLK, BBT), 256, 0, stream>>>(hbuf, attn, partials);
  k_head    <<<1, 256, 0, stream>>>(partials, Wv, ln1g, ln1b, gWih, gbih, gbhh,
                                    ln2g, ln2b, p1W, p1b, fbuf);
  k_pre2    <<<P2C/16/8, 256, 0, stream>>>(fbuf, p2W, p2b, nf);
  k_quant   <<<BBT*NND, 128, 0, stream>>>(nf, lW, lb, iW, ib, pred);
}
